// LinformerTransformerModel_85787676770464
// MI455X (gfx1250) — matrix operand order repaired and checked by reading, not compiled, not run on hardware
//
#include <hip/hip_runtime.h>

// ---------------------------------------------------------------------------
// Linformer forward for MI455X (gfx1250): bf16 WMMA GEMMs, fp32 accumulate.
// ---------------------------------------------------------------------------

typedef __bf16 bf16;
typedef bf16  v16bf __attribute__((ext_vector_type(16)));
typedef float v8f   __attribute__((ext_vector_type(8)));

#define DEV __device__ __forceinline__

namespace cfg {
constexpr int B = 16, N = 2048, IND = 64, DIM = 512, H = 8, L = 4, K = 256,
              NCLS = 2, DH = 64;
}

// ---------------------------------------------------------------------------
// WMMA fragment loaders (CDNA5 wave32 layouts, 16-bit data, 16x16x32)
// A is row-major [M,Kd]; lane holds row m=lane&15, 16 consecutive K values from (lane>>4)*16.
// B is supplied transposed (BT row-major [Nd,Kd]); lane holds col n=lane&15,
// 16 consecutive K values starting at (lane>>4)*16.
// ---------------------------------------------------------------------------
DEV v16bf load_a_frag(const bf16* A, long lda, int row0, int kk, int lane) {
  int m = lane & 15;
  int ko = (lane >> 4) * 16;
  const bf16* p = A + (long)(row0 + m) * lda + kk + ko;
  union { v16bf v; uint4 u[2]; } r;
  r.u[0] = *reinterpret_cast<const uint4*>(p);        // K = kb .. kb+7
  r.u[1] = *reinterpret_cast<const uint4*>(p + 8);    // K = kb+8 .. kb+15
  return r.v;
}

DEV v16bf load_b_frag(const bf16* Bt, long ldb, int col0, int kk, int lane) {
  int n = lane & 15;
  int ko = (lane >> 4) * 16;
  const bf16* p = Bt + (long)(col0 + n) * ldb + kk + ko;
  union { v16bf v; uint4 u[2]; } r;
  r.u[0] = *reinterpret_cast<const uint4*>(p);        // K = kb2 .. kb2+7
  r.u[1] = *reinterpret_cast<const uint4*>(p + 8);    // K = kb2+8 .. kb2+15
  return r.v;
}

DEV float gelu_tanh(float x) {
  float x3 = x * x * x;
  float t = tanhf(0.7978845608028654f * (x + 0.044715f * x3));
  return 0.5f * x * (1.0f + t);
}

// ---------------------------------------------------------------------------
// Generic batched GEMM: C[M,Nd] = act(alpha * A[M,Kd] @ B[Kd,Nd] + bias + resid)
// B given transposed (BT [Nd,Kd]). Batch index t = blockIdx.z, split into
// outer (t/innerB) and inner (t%innerB) strides for [B,H]-style batching.
// Block: 256 threads = 8 waves (4 row-groups x 2 col-groups), each wave owns a
// 32x64 C tile (2 A-frags x 4 B-frags = 8 WMMAs/k-step) -> 128x128 block tile.
// Software-pipelined: loads for step k+1 issue before WMMAs of step k.
// ---------------------------------------------------------------------------
__global__ void __launch_bounds__(256)
gemm_bf16_wmma(const bf16* __restrict__ A, long lda, long sAo, long sAi,
               const bf16* __restrict__ Bt, long ldb, long sBo, long sBi,
               int M, int Nd, int Kd, int innerB,
               const float* __restrict__ bias,
               const float* __restrict__ resid, long ldr, int residMod,
               float* __restrict__ Cf, long ldcf, long sCfo, long sCfi,
               bf16* __restrict__ Cb, long ldcb, long sCbo, long sCbi,
               float alpha, int act) {
  int t = blockIdx.z;
  int bo = t / innerB, bi = t - bo * innerB;
  A  += bo * sAo + bi * sAi;
  Bt += bo * sBo + bi * sBi;

  int wave = threadIdx.x >> 5, lane = threadIdx.x & 31;
  int row0 = blockIdx.x * 128 + (wave >> 1) * 32;
  int col0 = blockIdx.y * 128 + (wave & 1) * 64;
  if (row0 >= M || col0 >= Nd) return;   // wave-uniform: EXEC stays all-ones

  v8f acc[8];
#pragma unroll
  for (int i = 0; i < 8; ++i) acc[i] = (v8f){};

  // prologue: fragments for k-step 0
  v16bf a0 = load_a_frag(A, lda, row0, 0, lane);
  v16bf a1 = load_a_frag(A, lda, row0 + 16, 0, lane);
  v16bf b0 = load_b_frag(Bt, ldb, col0, 0, lane);
  v16bf b1 = load_b_frag(Bt, ldb, col0 + 16, 0, lane);
  v16bf b2 = load_b_frag(Bt, ldb, col0 + 32, 0, lane);
  v16bf b3 = load_b_frag(Bt, ldb, col0 + 48, 0, lane);

  for (int kk = 0; kk < Kd; kk += 32) {
    // prefetch next k-step (clamped in-bounds on the last iteration)
    int kn = (kk + 32 < Kd) ? (kk + 32) : kk;
    v16bf na0 = load_a_frag(A, lda, row0, kn, lane);
    v16bf na1 = load_a_frag(A, lda, row0 + 16, kn, lane);
    v16bf nb0 = load_b_frag(Bt, ldb, col0, kn, lane);
    v16bf nb1 = load_b_frag(Bt, ldb, col0 + 16, kn, lane);
    v16bf nb2 = load_b_frag(Bt, ldb, col0 + 32, kn, lane);
    v16bf nb3 = load_b_frag(Bt, ldb, col0 + 48, kn, lane);

    acc[0] = __builtin_amdgcn_wmma_f32_16x16x32_bf16(false, a0, false, b0,
                                                     (short)0, acc[0], false, false);
    acc[4] = __builtin_amdgcn_wmma_f32_16x16x32_bf16(false, a1, false, b0,
                                                     (short)0, acc[4], false, false);
    acc[1] = __builtin_amdgcn_wmma_f32_16x16x32_bf16(false, a0, false, b1,
                                                     (short)0, acc[1], false, false);
    acc[5] = __builtin_amdgcn_wmma_f32_16x16x32_bf16(false, a1, false, b1,
                                                     (short)0, acc[5], false, false);
    acc[2] = __builtin_amdgcn_wmma_f32_16x16x32_bf16(false, a0, false, b2,
                                                     (short)0, acc[2], false, false);
    acc[6] = __builtin_amdgcn_wmma_f32_16x16x32_bf16(false, a1, false, b2,
                                                     (short)0, acc[6], false, false);
    acc[3] = __builtin_amdgcn_wmma_f32_16x16x32_bf16(false, a0, false, b3,
                                                     (short)0, acc[3], false, false);
    acc[7] = __builtin_amdgcn_wmma_f32_16x16x32_bf16(false, a1, false, b3,
                                                     (short)0, acc[7], false, false);

    a0 = na0; a1 = na1; b0 = nb0; b1 = nb1; b2 = nb2; b3 = nb3;
  }

  // Epilogue. C layout: lane L -> col = L&15, rows = (L>>4)*8 + r (r=0..7).
  int cn = lane & 15;
  int rsel = (lane >> 4) * 8;
  const float* residB = resid;
  float* CfB = Cf ? Cf + bo * sCfo + bi * sCfi : nullptr;
  bf16*  CbB = Cb ? Cb + bo * sCbo + bi * sCbi : nullptr;

#pragma unroll
  for (int half = 0; half < 2; ++half) {
#pragma unroll
    for (int ch = 0; ch < 4; ++ch) {
      v8f& a = acc[half * 4 + ch];
      int rbase = row0 + half * 16 + rsel;
      int col = col0 + ch * 16 + cn;
      float bval = bias ? bias[col] : 0.0f;
#pragma unroll
      for (int r = 0; r < 8; ++r) {
        int row = rbase + r;
        float v = a[r] * alpha + bval;
        if (residB) {
          long rr = residMod ? (long)(row % residMod) : (long)row;
          v += residB[rr * ldr + col];
        }
        if (act == 1) v = gelu_tanh(v);
        if (CfB) CfB[(long)row * ldcf + col] = v;
        if (CbB) CbB[(long)row * ldcb + col] = (bf16)v;
      }
    }
  }
}

// ---------------------------------------------------------------------------
// Support kernels
// ---------------------------------------------------------------------------
__global__ void cvt_f32_bf16(const float* __restrict__ s, bf16* __restrict__ d,
                             long n) {
  long i = (long)blockIdx.x * blockDim.x + threadIdx.x;
  if (i < n) d[i] = (bf16)s[i];
}

__global__ void make_posb(const float* __restrict__ pos,
                          const float* __restrict__ emb_b,
                          float* __restrict__ posb, long n, int dim) {
  long i = (long)blockIdx.x * blockDim.x + threadIdx.x;
  if (i < n) posb[i] = pos[i] + emb_b[i % dim];
}

// dst[C,R] = transpose(src[R,C]), convert to bf16. block (32,8).
__global__ void transpose_f32_bf16(const float* __restrict__ src, long sB,
                                   bf16* __restrict__ dst, long dB, int R,
                                   int C) {
  __shared__ bf16 tile[32][33];
  const float* s = src + (long)blockIdx.z * sB;
  bf16* d = dst + (long)blockIdx.z * dB;
  int c0 = blockIdx.x * 32, r0 = blockIdx.y * 32;
  int tx = threadIdx.x, ty = threadIdx.y;
  for (int j = 0; j < 32; j += 8) {
    int r = r0 + ty + j, c = c0 + tx;
    tile[ty + j][tx] = (r < R && c < C) ? (bf16)s[(long)r * C + c] : (bf16)0.0f;
  }
  __syncthreads();
  for (int j = 0; j < 32; j += 8) {
    int c = c0 + ty + j, r = r0 + tx;
    if (c < C && r < R) d[(long)c * R + r] = tile[tx][ty + j];
  }
}

__global__ void transpose_bf16_bf16(const bf16* __restrict__ src, long sB,
                                    bf16* __restrict__ dst, long dB, int R,
                                    int C) {
  __shared__ bf16 tile[32][33];
  const bf16* s = src + (long)blockIdx.z * sB;
  bf16* d = dst + (long)blockIdx.z * dB;
  int c0 = blockIdx.x * 32, r0 = blockIdx.y * 32;
  int tx = threadIdx.x, ty = threadIdx.y;
  for (int j = 0; j < 32; j += 8) {
    int r = r0 + ty + j, c = c0 + tx;
    tile[ty + j][tx] = (r < R && c < C) ? s[(long)r * C + c] : (bf16)0.0f;
  }
  __syncthreads();
  for (int j = 0; j < 32; j += 8) {
    int c = c0 + ty + j, r = r0 + tx;
    if (c < C && r < R) d[(long)c * R + r] = tile[tx][ty + j];
  }
}

// One block (256 threads, 8 waves) per row.
__global__ void layernorm_bf16(const float* __restrict__ h,
                               const float* __restrict__ sc,
                               const float* __restrict__ bi,
                               bf16* __restrict__ z, int dim) {
  long row = blockIdx.x;
  const float* hr = h + row * dim;
  float sum = 0.f, sq = 0.f;
  for (int c = threadIdx.x; c < dim; c += blockDim.x) {
    float x = hr[c];
    sum += x;
    sq += x * x;
  }
  for (int o = 16; o; o >>= 1) {
    sum += __shfl_xor(sum, o, 32);
    sq += __shfl_xor(sq, o, 32);
  }
  __shared__ float s1[8], s2[8];
  int w = threadIdx.x >> 5;
  if ((threadIdx.x & 31) == 0) { s1[w] = sum; s2[w] = sq; }
  __syncthreads();
  float ts = 0.f, tq = 0.f;
  int nw = blockDim.x >> 5;
  for (int i = 0; i < nw; ++i) { ts += s1[i]; tq += s2[i]; }
  float mu = ts / dim;
  float var = tq / dim - mu * mu;
  float rs = rsqrtf(var + 1e-5f);
  bf16* zr = z + row * dim;
  for (int c = threadIdx.x; c < dim; c += blockDim.x)
    zr[c] = (bf16)((hr[c] - mu) * rs * sc[c] + bi[c]);
}

// One wave per row of K=256; 8 rows per 256-thread block.
__global__ void softmax_bf16(const float* __restrict__ dots,
                             bf16* __restrict__ attn) {
  constexpr int K = cfg::K;
  int wid = threadIdx.x >> 5, lane = threadIdx.x & 31;
  long row = (long)blockIdx.x * 8 + wid;
  const float* dr = dots + row * K;
  float v[K / 32];
  float m = -3.4e38f;
  for (int i = 0; i < K / 32; ++i) {
    v[i] = dr[lane + 32 * i];
    m = fmaxf(m, v[i]);
  }
  for (int o = 16; o; o >>= 1) m = fmaxf(m, __shfl_xor(m, o, 32));
  float s = 0.f;
  for (int i = 0; i < K / 32; ++i) {
    v[i] = __expf(v[i] - m);
    s += v[i];
  }
  for (int o = 16; o; o >>= 1) s += __shfl_xor(s, o, 32);
  float inv = 1.0f / s;
  bf16* ar = attn + row * K;
  for (int i = 0; i < K / 32; ++i) ar[lane + 32 * i] = (bf16)(v[i] * inv);
}

__global__ void meanpool(const float* __restrict__ h, float* __restrict__ pooled) {
  int idx = blockIdx.x * blockDim.x + threadIdx.x;
  if (idx >= cfg::B * cfg::DIM) return;
  int b = idx / cfg::DIM, d = idx % cfg::DIM;
  const float* p = h + (long)b * cfg::N * cfg::DIM + d;
  float s = 0.f;
  for (int n = 0; n < cfg::N; ++n) s += p[(long)n * cfg::DIM];
  pooled[idx] = s * (1.0f / cfg::N);
}

__global__ void fchead(const float* __restrict__ pooled,
                       const float* __restrict__ w,
                       const float* __restrict__ fb, float* __restrict__ out) {
  int idx = threadIdx.x;
  if (idx >= cfg::B * cfg::NCLS) return;
  int b = idx / cfg::NCLS, c = idx % cfg::NCLS;
  float s = fb[c];
  for (int d = 0; d < cfg::DIM; ++d)
    s += pooled[b * cfg::DIM + d] * w[d * cfg::NCLS + c];
  out[idx] = s;
}

// ---------------------------------------------------------------------------
// Host-side helpers
// ---------------------------------------------------------------------------
static void gemm(hipStream_t st, const bf16* A, long lda, long sAo, long sAi,
                 const bf16* Bt, long ldb, long sBo, long sBi, int M, int Nd,
                 int Kd, int batch, int innerB, const float* bias,
                 const float* resid, long ldr, int residMod, float* Cf,
                 long ldcf, long sCfo, long sCfi, bf16* Cb, long ldcb,
                 long sCbo, long sCbi, float alpha, int act) {
  dim3 g((unsigned)((M + 127) / 128), (unsigned)((Nd + 127) / 128),
         (unsigned)batch);
  gemm_bf16_wmma<<<g, 256, 0, st>>>(A, lda, sAo, sAi, Bt, ldb, sBo, sBi, M, Nd,
                                    Kd, innerB, bias, resid, ldr, residMod, Cf,
                                    ldcf, sCfo, sCfi, Cb, ldcb, sCbo, sCbi,
                                    alpha, act);
}

static void tr_f32(hipStream_t st, const float* s, bf16* d, int R, int C) {
  dim3 g((unsigned)((C + 31) / 32), (unsigned)((R + 31) / 32), 1);
  transpose_f32_bf16<<<g, dim3(32, 8), 0, st>>>(s, 0, d, 0, R, C);
}

static void tr_bf16_b(hipStream_t st, const bf16* s, long sB, bf16* d, long dB,
                      int R, int C, int batch) {
  dim3 g((unsigned)((C + 31) / 32), (unsigned)((R + 31) / 32), (unsigned)batch);
  transpose_bf16_bf16<<<g, dim3(32, 8), 0, st>>>(s, sB, d, dB, R, C);
}

extern "C" void kernel_launch(void* const* d_in, const int* in_sizes, int n_in,
                              void* d_out, int out_size, void* d_ws,
                              size_t ws_size, hipStream_t stream) {
  using namespace cfg;
  (void)in_sizes; (void)n_in; (void)out_size; (void)ws_size;

  const float* x     = (const float*)d_in[0];
  const float* emb_w = (const float*)d_in[1];
  const float* emb_b = (const float*)d_in[2];
  const float* pos   = (const float*)d_in[3];
  const float* ln1_s = (const float*)d_in[4];
  const float* ln1_b = (const float*)d_in[5];
  const float* wq    = (const float*)d_in[6];
  const float* wk    = (const float*)d_in[7];
  const float* wv    = (const float*)d_in[8];
  const float* pk    = (const float*)d_in[9];
  const float* pv    = (const float*)d_in[10];
  const float* wo    = (const float*)d_in[11];
  const float* wo_b  = (const float*)d_in[12];
  const float* ln2_s = (const float*)d_in[13];
  const float* ln2_b = (const float*)d_in[14];
  const float* w1    = (const float*)d_in[15];
  const float* b1    = (const float*)d_in[16];
  const float* w2    = (const float*)d_in[17];
  const float* b2    = (const float*)d_in[18];
  const float* fc_w  = (const float*)d_in[19];
  const float* fc_b  = (const float*)d_in[20];
  float* out = (float*)d_out;

  // ---- workspace arena -----------------------------------------------------
  char* ws = (char*)d_ws;
  size_t off = 0;
  auto carve = [&](size_t bytes) -> char* {
    char* p = ws + off;
    off = (off + bytes + 255) & ~(size_t)255;
    return p;
  };
  const size_t actE = (size_t)B * N * DIM;  // 16.7M elements
  float* h    = (float*)carve(actE * 4);
  float* posb = (float*)carve((size_t)N * DIM * 4);
  bf16* zb   = (bf16*)carve(actE * 2);
  bf16* qb   = (bf16*)carve(actE * 2);
  bf16* kv   = (bf16*)carve(actE * 2 * 2);  // kb|vb contiguous; dots aliases
  bf16* kb = kv;
  bf16* vb = kv + actE;
  bf16* tbuf = (bf16*)carve(actE * 2);  // kT / vT / ffn chunk tmp
  bf16* kpb  = (bf16*)carve((size_t)B * K * DIM * 2);
  bf16* vpb  = (bf16*)carve((size_t)B * K * DIM * 2);
  bf16* vpT  = (bf16*)carve((size_t)B * DIM * K * 2);
  bf16* xb   = (bf16*)carve((size_t)B * N * IND * 2);
  bf16* embT = (bf16*)carve((size_t)DIM * IND * 2);
  bf16* wTq  = (bf16*)carve((size_t)DIM * DIM * 2);
  bf16* wTk  = (bf16*)carve((size_t)DIM * DIM * 2);
  bf16* wTv  = (bf16*)carve((size_t)DIM * DIM * 2);
  bf16* wTo  = (bf16*)carve((size_t)DIM * DIM * 2);
  bf16* w1T  = (bf16*)carve((size_t)4 * DIM * DIM * 2);  // [2048,512]
  bf16* w2T  = (bf16*)carve((size_t)DIM * 4 * DIM * 2);  // [512,2048]
  bf16* pkT  = (bf16*)carve((size_t)K * N * 2);
  bf16* pvT  = (bf16*)carve((size_t)K * N * 2);
  float* pooled = (float*)carve((size_t)B * DIM * 4);
  // aliases (lifetimes verified: z dead after v; k,v dead after kp,vp; q dead
  // after dots; dots dead after softmax; attn dead after out-GEMM)
  float* dots    = (float*)kv;  // [B,H,N,K] f32 = 256MB = 4x(|kb|+|vb|)
  bf16* attn    = zb;           // [B,H,N,K] bf16
  bf16* attnout = qb;           // [B,N,DIM] bf16
  bf16* z2      = kb;           // [B,N,DIM] bf16
  bf16* ffT     = tbuf;         // [B*N, 512] bf16 ffn chunk

  const float scale = 0.125f;  // DH^-0.5 = 1/8
  const int Mrows = B * N;     // 32768

  // ---- embedding -----------------------------------------------------------
  long nx = (long)B * N * IND;
  cvt_f32_bf16<<<dim3((unsigned)((nx + 255) / 256)), 256, 0, stream>>>(x, xb, nx);
  tr_f32(stream, emb_w, embT, IND, DIM);  // -> [512,64]
  long np = (long)N * DIM;
  make_posb<<<dim3((unsigned)((np + 255) / 256)), 256, 0, stream>>>(pos, emb_b,
                                                                    posb, np, DIM);
  // h = x @ emb_w + (emb_b + pos[n])
  gemm(stream, xb, IND, 0, 0, embT, IND, 0, 0, Mrows, DIM, IND, 1, 1, nullptr,
       posb, DIM, N, h, DIM, 0, 0, nullptr, 0, 0, 0, 1.0f, 0);

  for (int i = 0; i < L; ++i) {
    const float* wq_i = wq + (size_t)i * DIM * DIM;
    const float* wk_i = wk + (size_t)i * DIM * DIM;
    const float* wv_i = wv + (size_t)i * DIM * DIM;
    const float* wo_i = wo + (size_t)i * DIM * DIM;
    const float* pk_i = pk + (size_t)i * N * K;
    const float* pv_i = pv + (size_t)i * N * K;
    const float* w1_i = w1 + (size_t)i * DIM * 4 * DIM;
    const float* w2_i = w2 + (size_t)i * 4 * DIM * DIM;

    // ---- attention block ----
    layernorm_bf16<<<dim3((unsigned)Mrows), 256, 0, stream>>>(
        h, ln1_s + (size_t)i * DIM, ln1_b + (size_t)i * DIM, zb, DIM);

    tr_f32(stream, wq_i, wTq, DIM, DIM);
    tr_f32(stream, wk_i, wTk, DIM, DIM);
    tr_f32(stream, wv_i, wTv, DIM, DIM);
    gemm(stream, zb, DIM, 0, 0, wTq, DIM, 0, 0, Mrows, DIM, DIM, 1, 1, nullptr,
         nullptr, 0, 0, nullptr, 0, 0, 0, qb, DIM, 0, 0, 1.0f, 0);
    gemm(stream, zb, DIM, 0, 0, wTk, DIM, 0, 0, Mrows, DIM, DIM, 1, 1, nullptr,
         nullptr, 0, 0, nullptr, 0, 0, 0, kb, DIM, 0, 0, 1.0f, 0);
    gemm(stream, zb, DIM, 0, 0, wTv, DIM, 0, 0, Mrows, DIM, DIM, 1, 1, nullptr,
         nullptr, 0, 0, nullptr, 0, 0, 0, vb, DIM, 0, 0, 1.0f, 0);

    tr_f32(stream, pk_i, pkT, N, K);  // [K,N]
    tr_f32(stream, pv_i, pvT, N, K);

    // kp[b] = pk^T @ k[b] : [256,2048] @ [2048,512]
    tr_bf16_b(stream, kb, (long)N * DIM, tbuf, (long)DIM * N, N, DIM, B);
    gemm(stream, pkT, N, 0, 0, tbuf, N, (long)DIM * N, 0, K, DIM, N, B, 1,
         nullptr, nullptr, 0, 0, nullptr, 0, 0, 0, kpb, DIM, (long)K * DIM, 0,
         1.0f, 0);
    tr_bf16_b(stream, vb, (long)N * DIM, tbuf, (long)DIM * N, N, DIM, B);
    gemm(stream, pvT, N, 0, 0, tbuf, N, (long)DIM * N, 0, K, DIM, N, B, 1,
         nullptr, nullptr, 0, 0, nullptr, 0, 0, 0, vpb, DIM, (long)K * DIM, 0,
         1.0f, 0);
    tr_bf16_b(stream, vpb, (long)K * DIM, vpT, (long)DIM * K, K, DIM, B);

    // dots[b,h] = scale * q[b,:,h,:] @ kp[b,:,h,:]^T   (batch = B*H)
    gemm(stream, qb, DIM, (long)N * DIM, DH, kpb, DIM, (long)K * DIM, DH, N, K,
         DH, B * H, H, nullptr, nullptr, 0, 0, dots, K, (long)H * N * K,
         (long)N * K, nullptr, 0, 0, 0, scale, 0);

    softmax_bf16<<<dim3((unsigned)(B * H * N / 8)), 256, 0, stream>>>(dots, attn);

    // out[b,:,h,:] = attn[b,h] @ vp[b,:,h,:]
    gemm(stream, attn, K, (long)H * N * K, (long)N * K, vpT, K, (long)DIM * K,
         (long)DH * K, N, DH, K, B * H, H, nullptr, nullptr, 0, 0, nullptr, 0,
         0, 0, attnout, DIM, (long)N * DIM, DH, 1.0f, 0);

    // h += out @ wo + wo_b
    tr_f32(stream, wo_i, wTo, DIM, DIM);
    gemm(stream, attnout, DIM, 0, 0, wTo, DIM, 0, 0, Mrows, DIM, DIM, 1, 1,
         wo_b + (size_t)i * DIM, h, DIM, 0, h, DIM, 0, 0, nullptr, 0, 0, 0,
         1.0f, 0);

    // ---- FFN block ----
    layernorm_bf16<<<dim3((unsigned)Mrows), 256, 0, stream>>>(
        h, ln2_s + (size_t)i * DIM, ln2_b + (size_t)i * DIM, z2, DIM);
    tr_f32(stream, w1_i, w1T, DIM, 4 * DIM);  // [2048,512]
    tr_f32(stream, w2_i, w2T, 4 * DIM, DIM);  // [512,2048]
    for (int c = 0; c < 4; ++c) {
      int c0 = c * DIM;
      // tmp = gelu(z2 @ w1[:, c0:c0+512] + b1[c0:])
      gemm(stream, z2, DIM, 0, 0, w1T + (size_t)c0 * DIM, DIM, 0, 0, Mrows,
           DIM, DIM, 1, 1, b1 + (size_t)i * 4 * DIM + c0, nullptr, 0, 0,
           nullptr, 0, 0, 0, ffT, DIM, 0, 0, 1.0f, 1);
      // h += tmp @ w2[c0:c0+512, :]  (+ b2 once)
      gemm(stream, ffT, DIM, 0, 0, w2T + c0, 4 * DIM, 0, 0, Mrows, DIM, DIM, 1,
           1, c == 0 ? (b2 + (size_t)i * DIM) : nullptr, h, DIM, 0, h, DIM, 0,
           0, nullptr, 0, 0, 0, 1.0f, 0);
    }
  }

  // ---- head ----------------------------------------------------------------
  meanpool<<<dim3((unsigned)((B * DIM + 255) / 256)), 256, 0, stream>>>(h,
                                                                        pooled);
  fchead<<<1, 32, 0, stream>>>(pooled, fc_w, fc_b, out);
}
